// MultiheadAttention_59880434040881
// MI455X (gfx1250) — matrix operand order repaired and checked by reading, not compiled, not run on hardware
//
#include <hip/hip_runtime.h>
#include <hip/hip_bf16.h>

#define NB 2
#define NS 2048
#define ND 512
#define NH 8
#define NDK 64

typedef __attribute__((ext_vector_type(16))) __bf16 v16bf;
typedef __attribute__((ext_vector_type(8)))  float  v8f;
typedef __attribute__((ext_vector_type(4)))  unsigned int u32x4;
typedef __attribute__((ext_vector_type(8)))  int i32x8;
typedef __attribute__((ext_vector_type(4)))  int i32x4;

#if defined(__has_builtin)
#if __has_builtin(__builtin_amdgcn_tensor_load_to_lds) && __has_builtin(__builtin_amdgcn_s_wait_tensorcnt)
#define USE_TDM 1
#endif
#endif
#ifndef USE_TDM
#define USE_TDM 0
#endif

__device__ __forceinline__ v8f wmma_bf16(v16bf a, v16bf b, v8f c) {
  // D = A(16x32 bf16) * B(32x16 bf16) + C(16x16 f32)
  return __builtin_amdgcn_wmma_f32_16x16x32_bf16(false, a, false, b, (short)0, c,
                                                 false, false);
}

// A-matrix (16x32, 16-bit) lane-element -> K index (ISA 7.12.2 table).
// lh = lane>>4 (half), e = element 0..15 within the 8-VGPR fragment.
__device__ __forceinline__ int a_kidx(int e, int lh) {
  int vg = e >> 1, p = e & 1;
  return (vg < 4) ? (8 * lh + 2 * vg + p) : (16 + 8 * lh + 2 * (vg - 4) + p);
}
// B-matrix (32x16, 16-bit): same K order as the A-matrix table.
__device__ __forceinline__ int b_kidx(int e, int lh) { return a_kidx(e, lh); }

#if USE_TDM
// Stage a [32 rows x 64 bf16] tile (contiguous rows, stride 64) into LDS via TDM.
// 6-arg builtin form: (g0 u32x4, g1 i32x8, g2 i32x4, g3 i32x4, ext i32x8, cpol i32).
__device__ __forceinline__ void tdm_load_tile(unsigned lds_off, const __bf16* gsrc) {
  unsigned long long ga = (unsigned long long)(size_t)(const void*)gsrc;
  u32x4 g0;
  g0[0] = 1u;                          // count=1 (valid user descriptor)
  g0[1] = lds_off;                     // lds_addr (bytes)
  g0[2] = (unsigned)ga;                // global_addr[31:0]
  g0[3] = (unsigned)((ga >> 32) & 0x1FFFFFFull) | (2u << 30); // addr[56:32] | type=2
  i32x8 g1;
  g1[0] = (1 << 16);     // data_size=1 (2 bytes/elem); workgroup_mask=0; no flags
  g1[1] = (64 << 16);    // tensor_dim0 = 64 (low16 at bits 63:48)
  g1[2] = (2048 << 16);  // tensor_dim1 = 2048 (low16 at bits 95:80)
  g1[3] = (64 << 16);    // tile_dim0 = 64 (bits 127:112)
  g1[4] = 32;            // tile_dim1 = 32 (bits 143:128)
  g1[5] = 64;            // tensor_dim0_stride = 64 (low32 of bits 207:160)
  g1[6] = 0;
  g1[7] = 0;
  i32x4 gz4 = {0, 0, 0, 0};                    // groups 2/3 unused for 2D tiles
  i32x8 gz8 = {0, 0, 0, 0, 0, 0, 0, 0};        // extra group (clang-23 arity)
  __builtin_amdgcn_tensor_load_to_lds(g0, g1, gz4, gz4, gz8, 0);
}
#endif

// ---------------------------------------------------------------------------
// QKV projection: out = (X @ W^T + b) * scale, written bf16 in [B,H,S,DK].
// Grid: (4096/16 row tiles, 512/64 col groups); 128 threads = 4 waves,
// one 16x16 output tile per wave, K-loop in steps of 32 -> 1 WMMA each.
// ---------------------------------------------------------------------------
__global__ void proj_gemm_bf16out(const float* __restrict__ X,
                                  const float* __restrict__ W,
                                  const float* __restrict__ bias,
                                  unsigned short* __restrict__ outp,
                                  float scale) {
  __bf16* out = (__bf16*)outp;
  int lane = threadIdx.x & 31, w = threadIdx.x >> 5;
  int lh = lane >> 4, lm = lane & 15;
  int row = blockIdx.x * 16 + lm;           // A-fragment row for this lane
  int col = blockIdx.y * 64 + w * 16 + lm;  // B column = output feature
  v8f acc = {0.f, 0.f, 0.f, 0.f, 0.f, 0.f, 0.f, 0.f};
  for (int kk = 0; kk < ND; kk += 32) {
    __builtin_prefetch(X + (size_t)row * ND + kk + 64, 0, 3);
    v16bf a, b;
#pragma unroll
    for (int e = 0; e < 16; ++e) {
      a[e] = (__bf16)X[(size_t)row * ND + kk + a_kidx(e, lh)];
      b[e] = (__bf16)W[(size_t)col * ND + kk + b_kidx(e, lh)];  // W^T: B[k][n]=W[n][k]
    }
    acc = wmma_bf16(a, b, acc);
  }
  float bv = bias[col];
#pragma unroll
  for (int i = 0; i < 8; ++i) {
    int m = i + 8 * lh;                 // C layout: VGPR i holds rows i / i+8
    int r = blockIdx.x * 16 + m;        // global row in [B*S]
    float v = (acc[i] + bv) * scale;
    int bb = r >> 11, s = r & (NS - 1);
    int h = col >> 6, dk = col & (NDK - 1);
    out[(((size_t)bb * NH + h) * NS + s) * NDK + dk] = (__bf16)v;
  }
}

// ---------------------------------------------------------------------------
// Output projection: out = X @ Wo^T + bo (fp32 out, row-major [B*S, D]).
// ---------------------------------------------------------------------------
__global__ void out_proj_gemm(const float* __restrict__ X,
                              const float* __restrict__ W,
                              const float* __restrict__ bias,
                              float* __restrict__ out) {
  int lane = threadIdx.x & 31, w = threadIdx.x >> 5;
  int lh = lane >> 4, lm = lane & 15;
  int row = blockIdx.x * 16 + lm;
  int col = blockIdx.y * 64 + w * 16 + lm;
  v8f acc = {0.f, 0.f, 0.f, 0.f, 0.f, 0.f, 0.f, 0.f};
  for (int kk = 0; kk < ND; kk += 32) {
    v16bf a, b;
#pragma unroll
    for (int e = 0; e < 16; ++e) {
      a[e] = (__bf16)X[(size_t)row * ND + kk + a_kidx(e, lh)];
      b[e] = (__bf16)W[(size_t)col * ND + kk + b_kidx(e, lh)];
    }
    acc = wmma_bf16(a, b, acc);
  }
  float bv = bias[col];
#pragma unroll
  for (int i = 0; i < 8; ++i) {
    int m = i + 8 * lh;
    int r = blockIdx.x * 16 + m;
    out[(size_t)r * ND + col] = acc[i] + bv;
  }
}

// ---------------------------------------------------------------------------
// Flash attention, causal. Block = 64 query rows of one (b,h); 4 waves, each
// owning a 16-row Q tile. 32-key tiles staged to LDS (TDM when available).
// Online softmax in fp32; scores & P*V via bf16 WMMA. Q pre-scaled by 1/8.
// ---------------------------------------------------------------------------
__global__ void flash_attn(const unsigned short* __restrict__ Qp,
                           const unsigned short* __restrict__ Kp,
                           const unsigned short* __restrict__ Vp,
                           float* __restrict__ O) {
  const __bf16* Q = (const __bf16*)Qp;
  const __bf16* K = (const __bf16*)Kp;
  const __bf16* V = (const __bf16*)Vp;
  __shared__ __bf16 sK[32 * NDK];     // 4 KB key tile
  __shared__ __bf16 sV[32 * NDK];     // 4 KB value tile
  __shared__ __bf16 sP[4][16 * 32];   // per-wave P re-striping buffer

  const v8f vz = {0.f, 0.f, 0.f, 0.f, 0.f, 0.f, 0.f, 0.f};
  int lane = threadIdx.x & 31, w = threadIdx.x >> 5;
  int lh = lane >> 4, lm = lane & 15;
  int bh = blockIdx.y;
  int qblock = blockIdx.x * 64;
  int qwave = qblock + w * 16;

  // Q fragments for this wave: 16 rows x 64 dk, as two 16x32 A fragments.
  const __bf16* qrow = Q + ((size_t)bh * NS + qwave + lm) * NDK;
  v16bf Aq0, Aq1;
#pragma unroll
  for (int e = 0; e < 16; ++e) {
    Aq0[e] = qrow[a_kidx(e, lh)];
    Aq1[e] = qrow[32 + a_kidx(e, lh)];
  }

  v8f Oacc[4];
#pragma unroll
  for (int c = 0; c < 4; ++c) Oacc[c] = vz;
  float mrun[8], lrun[8];
#pragma unroll
  for (int i = 0; i < 8; ++i) { mrun[i] = -3.0e38f; lrun[i] = 0.f; }

  int kend = qblock + 64;  // causal: no keys beyond the block's last query
  for (int kt = 0; kt < kend; kt += 32) {
    __syncthreads();  // previous tile's LDS consumers done
    const __bf16* gk = K + ((size_t)bh * NS + kt) * NDK;
    const __bf16* gv = V + ((size_t)bh * NS + kt) * NDK;
#if USE_TDM
    if (threadIdx.x < 32) {  // wave 0 drives the Tensor Data Mover
      tdm_load_tile((unsigned)(size_t)(void*)sK, gk);
      tdm_load_tile((unsigned)(size_t)(void*)sV, gv);
      __builtin_amdgcn_s_wait_tensorcnt((short)0);
    }
#else
    for (int t = threadIdx.x; t < (32 * NDK) / 8; t += blockDim.x) {
      ((uint4*)sK)[t] = ((const uint4*)gk)[t];
      ((uint4*)sV)[t] = ((const uint4*)gv)[t];
    }
#endif
    __syncthreads();

    // ---- scores: S[16 q x 32 k] = Q * K^T (4 WMMAs) ----
    v16bf b0a, b0b, b1a, b1b;
#pragma unroll
    for (int e = 0; e < 16; ++e) {
      int k0 = b_kidx(e, lh);
      b0a[e] = sK[lm * NDK + k0];             // keys kt..kt+15, dk 0..31
      b0b[e] = sK[lm * NDK + 32 + k0];        // keys kt..kt+15, dk 32..63
      b1a[e] = sK[(16 + lm) * NDK + k0];      // keys kt+16..kt+31
      b1b[e] = sK[(16 + lm) * NDK + 32 + k0];
    }
    v8f S0 = vz, S1 = vz;
    S0 = wmma_bf16(Aq0, b0a, S0);
    S0 = wmma_bf16(Aq1, b0b, S0);
    S1 = wmma_bf16(Aq0, b1a, S1);
    S1 = wmma_bf16(Aq1, b1b, S1);

    // ---- causal mask + online softmax (rows live in 16-lane halves) ----
    float alpha[8];
#pragma unroll
    for (int i = 0; i < 8; ++i) {
      int m = i + 8 * lh;
      int q = qwave + m;
      float s0 = (kt + lm <= q) ? S0[i] : -1.0e9f;
      float s1 = (kt + 16 + lm <= q) ? S1[i] : -1.0e9f;
      float v = fmaxf(s0, s1);
      v = fmaxf(v, __shfl_xor(v, 1, 32));
      v = fmaxf(v, __shfl_xor(v, 2, 32));
      v = fmaxf(v, __shfl_xor(v, 4, 32));
      v = fmaxf(v, __shfl_xor(v, 8, 32));
      float mn = fmaxf(mrun[i], v);
      float a = __expf(mrun[i] - mn);
      float p0 = __expf(s0 - mn);
      float p1 = __expf(s1 - mn);
      float rs = p0 + p1;
      rs += __shfl_xor(rs, 1, 32);
      rs += __shfl_xor(rs, 2, 32);
      rs += __shfl_xor(rs, 4, 32);
      rs += __shfl_xor(rs, 8, 32);
      lrun[i] = lrun[i] * a + rs;
      mrun[i] = mn;
      alpha[i] = a;
      // C-layout -> LDS so we can re-stripe P into A-layout (same-wave, in-order DS)
      sP[w][m * 32 + lm] = (__bf16)p0;
      sP[w][m * 32 + 16 + lm] = (__bf16)p1;
    }
#pragma unroll
    for (int c = 0; c < 4; ++c)
#pragma unroll
      for (int i = 0; i < 8; ++i) Oacc[c][i] *= alpha[i];

    // ---- O += P(16x32) * V(32x64) (4 WMMAs) ----
    v16bf Ap;
#pragma unroll
    for (int e = 0; e < 16; ++e) Ap[e] = sP[w][lm * 32 + a_kidx(e, lh)];
#pragma unroll
    for (int c = 0; c < 4; ++c) {
      v16bf bvf;
#pragma unroll
      for (int e = 0; e < 16; ++e) bvf[e] = sV[b_kidx(e, lh) * NDK + c * 16 + lm];
      Oacc[c] = wmma_bf16(Ap, bvf, Oacc[c]);
    }
  }

  // ---- epilogue: normalize by l, write merged-head fp32 [B,S,D] ----
  int h = bh & (NH - 1), bb = bh >> 3;
#pragma unroll
  for (int c = 0; c < 4; ++c)
#pragma unroll
    for (int i = 0; i < 8; ++i) {
      int m = i + 8 * lh;
      int srow = qwave + m;
      int dcol = h * NDK + c * 16 + lm;
      O[((size_t)bb * NS + srow) * ND + dcol] = Oacc[c][i] / lrun[i];
    }
}

// ---------------------------------------------------------------------------
// Host launch. Workspace: Qbf(4MiB) | Kbf(4MiB) | Vbf(4MiB) | O f32(8MiB).
// ---------------------------------------------------------------------------
extern "C" void kernel_launch(void* const* d_in, const int* in_sizes, int n_in,
                              void* d_out, int out_size, void* d_ws, size_t ws_size,
                              hipStream_t stream) {
  (void)in_sizes; (void)n_in; (void)out_size; (void)ws_size;
  const float* query = (const float*)d_in[0];
  const float* key   = (const float*)d_in[1];
  const float* value = (const float*)d_in[2];
  // d_in[3] is the causal mask (int32); reproduced analytically in-kernel.
  const float* Wq = (const float*)d_in[4];
  const float* bq = (const float*)d_in[5];
  const float* Wk = (const float*)d_in[6];
  const float* bk = (const float*)d_in[7];
  const float* Wv = (const float*)d_in[8];
  const float* bv = (const float*)d_in[9];
  const float* Wo = (const float*)d_in[10];
  const float* bo = (const float*)d_in[11];

  char* ws = (char*)d_ws;
  unsigned short* Qbf = (unsigned short*)(ws + 0);
  unsigned short* Kbf = (unsigned short*)(ws + ((size_t)4 << 20));
  unsigned short* Vbf = (unsigned short*)(ws + ((size_t)8 << 20));
  float* Obuf = (float*)(ws + ((size_t)12 << 20));

  dim3 gProj(256, 8);  // (B*S)/16 row tiles x D/64 col groups
  dim3 blk(128);       // 4 wave32
  const float qscale = 0.125f;  // 1/sqrt(DK=64), folded into Q
  proj_gemm_bf16out<<<gProj, blk, 0, stream>>>(query, Wq, bq, Qbf, qscale);
  proj_gemm_bf16out<<<gProj, blk, 0, stream>>>(key,   Wk, bk, Kbf, 1.0f);
  proj_gemm_bf16out<<<gProj, blk, 0, stream>>>(value, Wv, bv, Vbf, 1.0f);
  flash_attn<<<dim3(NS / 64, NB * NH), blk, 0, stream>>>(Qbf, Kbf, Vbf, Obuf);
  out_proj_gemm<<<gProj, blk, 0, stream>>>(Obuf, Wo, bo, (float*)d_out);
}
